// SutraV05_74380243632416
// MI455X (gfx1250) — matrix operand order repaired and checked by reading, not compiled, not run on hardware
//
#include <hip/hip_runtime.h>
#include <hip/hip_bf16.h>

// ---------------- problem constants ----------------
#define B_    8
#define T_    4096
#define P_    4
#define D_    512
#define V_    256
#define N_    1024          // T/P
#define KTOP  16
#define W_    4
#define H_    8
#define HD_   64
#define ROUNDS_ 6
#define MN_   8192          // B*N

typedef __attribute__((ext_vector_type(16))) _Float16 v16h;
typedef __attribute__((ext_vector_type(8)))  _Float16 v8h;
typedef __attribute__((ext_vector_type(4)))  _Float16 v4h;
typedef __attribute__((ext_vector_type(8)))  float    v8f;
typedef __attribute__((ext_vector_type(4)))  float    f4;
typedef __attribute__((ext_vector_type(2)))  float    f2;
typedef int v4i_vs __attribute__((vector_size(16)));   // matches clang builtin V4i

union F16Frag { v16h v; v8h h[2]; };

#if defined(__has_builtin)
#if __has_builtin(__builtin_amdgcn_global_load_async_to_lds_b128) && \
    __has_builtin(__builtin_amdgcn_s_wait_asynccnt)
#define SUTRA_ASYNC_LDS 1
#endif
#endif

// ---------------- WMMA GEMM ----------------
// Tile: 128(M) x 64(N), K-step 32. 256 threads = 8 waves; wave w owns rows
// [16w,16w+16), 4 WMMA fragments across N. A/B are f32 in memory, converted
// to f16 while staging into LDS. Register-level double buffering: next K-tile
// is loaded to registers while WMMA consumes the current LDS tile.
// REQUIRES: M % 128 == 0, Nn % 64 == 0, Kk % 32 == 0 (all call sites comply).
#define TM 128
#define TN 64
#define TK 32
#define LDA_S 40   // 32 + 8 halves pad (80B rows -> 16B aligned)
#define LDB_S 40   // lB stored [n][k], 32 + 8 pad

__global__ __launch_bounds__(256)
void sutra_gemm_ab(const float* __restrict__ A, long lda, long sA,
                   const float* __restrict__ Bw, long ldb, long sB,
                   float* __restrict__ C, long ldc, long sC,
                   const float* __restrict__ bias,
                   int M, int Nn, int Kk, float alpha, int accum)
{
    __shared__ __align__(16) _Float16 lA[TM * LDA_S];
    __shared__ __align__(16) _Float16 lB[TN * LDB_S];

    const int tid  = threadIdx.x;
    const int wv   = tid >> 5;
    const int lane = tid & 31;
    const int half = lane >> 4;
    const int l16  = lane & 15;
    const int bm   = blockIdx.y * TM;
    const int bn   = blockIdx.x * TN;
    const int z    = blockIdx.z;

    const float* Ab = A  + (size_t)z * sA + (size_t)bm * lda;
    const float* Bb = Bw + (size_t)z * sB + bn;
    float*       Cb = C  + (size_t)z * sC;

    // A staging: thread -> row rA = tid>>1 (0..127), col base cA = (tid&1)*16
    const int rA = tid >> 1, cA = (tid & 1) * 16;
    // B staging: 4(k) x 2(n) sub-block per thread: kb = (tid>>5)*4, nb = (tid&31)*2
    const int kb = (tid >> 5) * 4, nb = (tid & 31) * 2;

    f4 areg[4];
    f2 breg[4];
    {   // prologue: load K-tile 0 into registers
        const float* ap = Ab + (size_t)rA * lda + cA;
        areg[0] = *(const f4*)(const void*)(ap);
        areg[1] = *(const f4*)(const void*)(ap + 4);
        areg[2] = *(const f4*)(const void*)(ap + 8);
        areg[3] = *(const f4*)(const void*)(ap + 12);
        const float* bp = Bb + (size_t)kb * ldb + nb;
        breg[0] = *(const f2*)(const void*)(bp);
        breg[1] = *(const f2*)(const void*)(bp + ldb);
        breg[2] = *(const f2*)(const void*)(bp + 2 * (size_t)ldb);
        breg[3] = *(const f2*)(const void*)(bp + 3 * (size_t)ldb);
    }

    v8f acc[4] = {};

    for (int k0 = 0; k0 < Kk; k0 += TK) {
        // ---- commit staged registers to LDS (convert f32 -> f16) ----
        {
            v8h h0, h1;
#pragma unroll
            for (int i = 0; i < 4; ++i) {
                h0[i]     = (_Float16)areg[0][i];
                h0[4 + i] = (_Float16)areg[1][i];
                h1[i]     = (_Float16)areg[2][i];
                h1[4 + i] = (_Float16)areg[3][i];
            }
            *(v8h*)(void*)&lA[rA * LDA_S + cA]     = h0;
            *(v8h*)(void*)&lA[rA * LDA_S + cA + 8] = h1;
#pragma unroll
            for (int j = 0; j < 2; ++j) {          // transpose 4x2 in registers
                v4h q;
                q[0] = (_Float16)breg[0][j];
                q[1] = (_Float16)breg[1][j];
                q[2] = (_Float16)breg[2][j];
                q[3] = (_Float16)breg[3][j];
                *(v4h*)(void*)&lB[(nb + j) * LDB_S + kb] = q;
            }
        }
        __syncthreads();

        // ---- issue next K-tile global loads (latency hidden by WMMA below) ----
        if (k0 + TK < Kk) {
            const float* ap = Ab + (size_t)rA * lda + (k0 + TK) + cA;
            areg[0] = *(const f4*)(const void*)(ap);
            areg[1] = *(const f4*)(const void*)(ap + 4);
            areg[2] = *(const f4*)(const void*)(ap + 8);
            areg[3] = *(const f4*)(const void*)(ap + 12);
            const float* bp = Bb + (size_t)(k0 + TK + kb) * ldb + nb;
            breg[0] = *(const f2*)(const void*)(bp);
            breg[1] = *(const f2*)(const void*)(bp + ldb);
            breg[2] = *(const f2*)(const void*)(bp + 2 * (size_t)ldb);
            breg[3] = *(const f2*)(const void*)(bp + 3 * (size_t)ldb);
        }

        // ---- WMMA on current LDS tile ----
        F16Frag af;
        const int ar = wv * 16 + l16;
        af.h[0] = *(const v8h*)(const void*)&lA[ar * LDA_S + half * 8];
        af.h[1] = *(const v8h*)(const void*)&lA[ar * LDA_S + 16 + half * 8];
#pragma unroll
        for (int j = 0; j < 4; ++j) {
            F16Frag bf;
            const int bc = j * 16 + l16;
            bf.h[0] = *(const v8h*)(const void*)&lB[bc * LDB_S + half * 8];
            bf.h[1] = *(const v8h*)(const void*)&lB[bc * LDB_S + 16 + half * 8];
            acc[j] = __builtin_amdgcn_wmma_f32_16x16x32_f16(
                false, af.v, false, bf.v, (short)0, acc[j], false, false);
        }
        __syncthreads();
    }

    // C layout: VGPR r -> row r (lanes 0-15) / row 8+r (lanes 16-31), col = lane&15
#pragma unroll
    for (int j = 0; j < 4; ++j) {
        int col = bn + j * 16 + l16;
        float bv = bias ? bias[col] : 0.f;
#pragma unroll
        for (int r = 0; r < 8; ++r) {
            int row = bm + wv * 16 + r + half * 8;
            size_t off = (size_t)row * ldc + col;
            float v = alpha * acc[j][r] + bv;
            if (accum) v += Cb[off];
            Cb[off] = v;
        }
    }
}

// C = mask_causal( alpha * A(MxK) @ B(NxK)^T ), invalid -> -inf. Batched.
// REQUIRES: M % 128 == 0, Nn % 64 == 0, Kk % 32 == 0.
__global__ __launch_bounds__(256)
void sutra_gemm_abt_causal(const float* __restrict__ A, long lda, long sA,
                           const float* __restrict__ Bt, long ldb, long sB,
                           float* __restrict__ C, long ldc, long sC,
                           int M, int Nn, int Kk, float alpha)
{
    __shared__ __align__(16) _Float16 lA[TM * LDA_S];
    __shared__ __align__(16) _Float16 lB[TN * LDB_S];

    const int tid  = threadIdx.x;
    const int wv   = tid >> 5;
    const int lane = tid & 31;
    const int half = lane >> 4;
    const int l16  = lane & 15;
    const int bm   = blockIdx.y * TM;
    const int bn   = blockIdx.x * TN;
    const int z    = blockIdx.z;

    const float* Ab = A  + (size_t)z * sA + (size_t)bm * lda;
    const float* Bb = Bt + (size_t)z * sB + (size_t)bn * ldb;
    float*       Cb = C  + (size_t)z * sC;

    v8f acc[4] = {};
    const float NEGINF = -__builtin_inff();

    if (bn <= bm + TM - 1) {   // tile intersects the causal region
        // A staging: row rA = tid>>1, col base cA = (tid&1)*16
        const int rA = tid >> 1, cA = (tid & 1) * 16;
        // B staging (B is NxK row-major, LDS wants [n][k]: direct copy):
        // thread -> n = tid>>2 (0..63), k base kc = (tid&3)*8
        const int nB = tid >> 2, kc = (tid & 3) * 8;

        f4 areg[4], breg[2];
        {
            const float* ap = Ab + (size_t)rA * lda + cA;
            areg[0] = *(const f4*)(const void*)(ap);
            areg[1] = *(const f4*)(const void*)(ap + 4);
            areg[2] = *(const f4*)(const void*)(ap + 8);
            areg[3] = *(const f4*)(const void*)(ap + 12);
            const float* bp = Bb + (size_t)nB * ldb + kc;
            breg[0] = *(const f4*)(const void*)(bp);
            breg[1] = *(const f4*)(const void*)(bp + 4);
        }

        for (int k0 = 0; k0 < Kk; k0 += TK) {
            {
                v8h h0, h1, hb;
#pragma unroll
                for (int i = 0; i < 4; ++i) {
                    h0[i]     = (_Float16)areg[0][i];
                    h0[4 + i] = (_Float16)areg[1][i];
                    h1[i]     = (_Float16)areg[2][i];
                    h1[4 + i] = (_Float16)areg[3][i];
                    hb[i]     = (_Float16)breg[0][i];
                    hb[4 + i] = (_Float16)breg[1][i];
                }
                *(v8h*)(void*)&lA[rA * LDA_S + cA]     = h0;
                *(v8h*)(void*)&lA[rA * LDA_S + cA + 8] = h1;
                *(v8h*)(void*)&lB[nB * LDB_S + kc]     = hb;
            }
            __syncthreads();

            if (k0 + TK < Kk) {
                const float* ap = Ab + (size_t)rA * lda + (k0 + TK) + cA;
                areg[0] = *(const f4*)(const void*)(ap);
                areg[1] = *(const f4*)(const void*)(ap + 4);
                areg[2] = *(const f4*)(const void*)(ap + 8);
                areg[3] = *(const f4*)(const void*)(ap + 12);
                const float* bp = Bb + (size_t)nB * ldb + (k0 + TK) + kc;
                breg[0] = *(const f4*)(const void*)(bp);
                breg[1] = *(const f4*)(const void*)(bp + 4);
            }

            F16Frag af;
            const int ar = wv * 16 + l16;
            af.h[0] = *(const v8h*)(const void*)&lA[ar * LDA_S + half * 8];
            af.h[1] = *(const v8h*)(const void*)&lA[ar * LDA_S + 16 + half * 8];
#pragma unroll
            for (int j = 0; j < 4; ++j) {
                F16Frag bf;
                const int bc = j * 16 + l16;
                bf.h[0] = *(const v8h*)(const void*)&lB[bc * LDB_S + half * 8];
                bf.h[1] = *(const v8h*)(const void*)&lB[bc * LDB_S + 16 + half * 8];
                acc[j] = __builtin_amdgcn_wmma_f32_16x16x32_f16(
                    false, af.v, false, bf.v, (short)0, acc[j], false, false);
            }
            __syncthreads();
        }
    }

#pragma unroll
    for (int j = 0; j < 4; ++j) {
        int col = bn + j * 16 + l16;
#pragma unroll
        for (int r = 0; r < 8; ++r) {
            int row = bm + wv * 16 + r + half * 8;
            float v = (col <= row) ? alpha * acc[j][r] : NEGINF;
            Cb[(size_t)row * ldc + col] = v;
        }
    }
}

// ---------------- elementwise / reduction kernels ----------------

__global__ void sutra_rope_table(float* __restrict__ cosT, float* __restrict__ sinT)
{
    int idx = blockIdx.x * blockDim.x + threadIdx.x;
    if (idx >= N_ * 256) return;
    int n = idx >> 8, i = idx & 255;
    float f = (float)n * __expf(-(float)i * (9.210340371976184f / 256.f)); // ln(1e4)
    cosT[idx] = __cosf(f);
    sinT[idx] = __sinf(f);
}

__global__ void sutra_embed(const int* __restrict__ x, const float* __restrict__ emb,
                            const float* __restrict__ pos, float* __restrict__ h0)
{
    int idx = blockIdx.x * blockDim.x + threadIdx.x;
    if (idx >= B_ * N_ * P_ * D_) return;
    int d = idx & 511;
    int p = (idx >> 9) & 3;
    int n = (idx >> 11) & 1023;
    int b = idx >> 21;
    int tok = x[b * T_ + n * P_ + p];
    h0[idx] = emb[tok * D_ + d] + pos[p * D_ + d];
}

__global__ void sutra_gru_gate(const float* __restrict__ gi, const float* __restrict__ gh,
                               const float* __restrict__ hprev, long hstride,
                               float* __restrict__ out, long ostride, int Mrows)
{
    int idx = blockIdx.x * blockDim.x + threadIdx.x;
    if (idx >= Mrows * D_) return;
    int m = idx >> 9, d = idx & 511;
    const float* gim = gi + (size_t)m * 1536;
    const float* ghm = gh + (size_t)m * 1536;
    float r  = 1.f / (1.f + __expf(-(gim[d]        + ghm[d])));
    float z  = 1.f / (1.f + __expf(-(gim[d + 512]  + ghm[d + 512])));
    float nn = tanhf(gim[d + 1024] + r * ghm[d + 1024]);
    float hp = hprev[(size_t)m * hstride + d];
    out[(size_t)m * ostride + d] = (1.f - z) * nn + z * hp;
}

__global__ __launch_bounds__(256)
void sutra_layernorm(const float* __restrict__ in, long istride,
                     float* __restrict__ out, long ostride,
                     const float* __restrict__ g, const float* __restrict__ b, int rows)
{
    int row = blockIdx.x;
    if (row >= rows) return;
    const float* x = in  + (size_t)row * istride;
    float*       o = out + (size_t)row * ostride;
    __shared__ float red[256];
    int tid = threadIdx.x;
    float a0 = x[tid], a1 = x[tid + 256];
    red[tid] = a0 + a1; __syncthreads();
    for (int st = 128; st > 0; st >>= 1) { if (tid < st) red[tid] += red[tid + st]; __syncthreads(); }
    float mu = red[0] * (1.f / 512.f); __syncthreads();
    float d0 = a0 - mu, d1 = a1 - mu;
    red[tid] = d0 * d0 + d1 * d1; __syncthreads();
    for (int st = 128; st > 0; st >>= 1) { if (tid < st) red[tid] += red[tid + st]; __syncthreads(); }
    float rstd = rsqrtf(red[0] * (1.f / 512.f) + 1e-5f);
    o[tid]       = d0 * rstd * g[tid]       + b[tid];
    o[tid + 256] = d1 * rstd * g[tid + 256] + b[tid + 256];
}

__global__ void sutra_patch_mean(const float* __restrict__ local, float* __restrict__ tmp)
{
    int idx = blockIdx.x * blockDim.x + threadIdx.x;
    if (idx >= MN_ * D_) return;
    int m = idx >> 9, d = idx & 511;
    const float* lm = local + (size_t)m * 2048 + d;
    tmp[idx] = 0.25f * (lm[0] + lm[512] + lm[1024] + lm[1536]);
}

__global__ void sutra_rotary(float* __restrict__ t, const float* __restrict__ cosT,
                             const float* __restrict__ sinT)
{
    int idx = blockIdx.x * blockDim.x + threadIdx.x;
    if (idx >= B_ * N_ * 256) return;
    int row = idx >> 8, i = idx & 255;
    int n = row & 1023;
    float c = cosT[n * 256 + i], s = sinT[n * 256 + i];
    float* p = t + (size_t)row * 512;
    float t1 = p[i], t2 = p[256 + i];
    p[i]       = t1 * c - t2 * s;
    p[256 + i] = t2 * c + t1 * s;
}

// G = mean_{w=0..4} gelu(U + shift_w(V) + b1)
__global__ void sutra_msg_gelu(const float* __restrict__ U, const float* __restrict__ Vw,
                               const float* __restrict__ b1, float* __restrict__ G)
{
    int idx = blockIdx.x * blockDim.x + threadIdx.x;
    if (idx >= MN_ * D_) return;
    int m = idx >> 9, d = idx & 511;
    int n = m & 1023;
    float u = U[idx] + b1[d];
    float acc = 0.f;
#pragma unroll
    for (int w = 0; w <= W_; ++w) {
        float nv = (n >= w) ? Vw[idx - (size_t)w * 512] : 0.f;
        float xg = u + nv;
        acc += 0.5f * xg * (1.f + erff(xg * 0.7071067811865475f));
    }
    G[idx] = acc * 0.2f;
}

// top-16 of a causal score row + softmax + gather-weighted sum of v
__global__ __launch_bounds__(256)
void sutra_topk_attend(const float* __restrict__ scores, const float* __restrict__ v,
                       float* __restrict__ attn)
{
    int m = blockIdx.x;                 // 0..MN-1
    int b = m >> 10, n = m & 1023;
    const float* srow = scores + (size_t)b * N_ * N_ + (size_t)n * N_;
    __shared__ __align__(16) float vals[N_];
    __shared__ float rv[256];
    __shared__ int   ri[256];
    __shared__ float bv[KTOP];
    __shared__ int   bi[KTOP];
    int tid = threadIdx.x;
    const float NEGINF = -__builtin_inff();
#ifdef SUTRA_ASYNC_LDS
    // CDNA5 async copy: each of 256 lanes moves 16B -> whole 4KB row in one op
    __builtin_amdgcn_global_load_async_to_lds_b128(
        (__attribute__((address_space(1))) v4i_vs*)(srow + tid * 4),
        (__attribute__((address_space(3))) v4i_vs*)&vals[tid * 4], 0, 0);
    __builtin_amdgcn_s_wait_asynccnt(0);
    __syncthreads();
#else
    for (int j = tid; j < N_; j += 256) vals[j] = srow[j];
    __syncthreads();
#endif
    for (int k = 0; k < KTOP; ++k) {
        float best = NEGINF; int bidx = 0;
        for (int j = tid; j < N_; j += 256)
            if (vals[j] > best) { best = vals[j]; bidx = j; }
        rv[tid] = best; ri[tid] = bidx; __syncthreads();
        for (int st = 128; st > 0; st >>= 1) {
            if (tid < st && rv[tid + st] > rv[tid]) { rv[tid] = rv[tid + st]; ri[tid] = ri[tid + st]; }
            __syncthreads();
        }
        if (tid == 0) { bv[k] = rv[0]; bi[k] = ri[0]; vals[ri[0]] = NEGINF; }
        __syncthreads();
    }
    float mx = bv[0];
    float p[KTOP]; float sum = 0.f;
#pragma unroll
    for (int k = 0; k < KTOP; ++k) {
        float e = (bv[k] == NEGINF) ? 0.f : __expf(bv[k] - mx);
        p[k] = e; sum += e;
    }
    float inv = 1.f / sum;
    for (int d = tid; d < D_; d += 256) {
        float acc = 0.f;
#pragma unroll
        for (int k = 0; k < KTOP; ++k)
            acc += p[k] * v[((size_t)b * N_ + bi[k]) * D_ + d];
        attn[(size_t)m * D_ + d] = acc * inv;
    }
}

__global__ __launch_bounds__(256)
void sutra_softmax_row(float* __restrict__ s, int len)
{
    int row = blockIdx.x;
    float* x = s + (size_t)row * len;
    __shared__ float red[256];
    int tid = threadIdx.x;
    float mx = -__builtin_inff();
    for (int j = tid; j < len; j += 256) mx = fmaxf(mx, x[j]);
    red[tid] = mx; __syncthreads();
    for (int st = 128; st > 0; st >>= 1) { if (tid < st) red[tid] = fmaxf(red[tid], red[tid + st]); __syncthreads(); }
    mx = red[0]; __syncthreads();
    float sm = 0.f;
    for (int j = tid; j < len; j += 256) { float e = __expf(x[j] - mx); x[j] = e; sm += e; }
    red[tid] = sm; __syncthreads();
    for (int st = 128; st > 0; st >>= 1) { if (tid < st) red[tid] += red[tid + st]; __syncthreads(); }
    float inv = 1.f / red[0];
    for (int j = tid; j < len; j += 256) x[j] *= inv;
}

__global__ void sutra_final_add(float* __restrict__ local, const float* __restrict__ broad)
{
    int idx = blockIdx.x * blockDim.x + threadIdx.x;
    if (idx >= B_ * T_ * D_) return;
    int d = idx & 511;
    int t = (idx >> 9) & 4095;
    int b = idx >> 21;
    local[idx] += broad[((size_t)b * N_ + (t >> 2)) * D_ + d];
}

// ---------------- host side ----------------
static inline int cdiv_(int a, int b) { return (a + b - 1) / b; }

static void G_AB(hipStream_t st, const float* A, long lda, long sA,
                 const float* Bw, long ldb, long sB,
                 float* C, long ldc, long sC, const float* bias,
                 int M, int Nn, int Kk, float alpha, int accum, int batch)
{
    dim3 g(cdiv_(Nn, TN), cdiv_(M, TM), batch);
    sutra_gemm_ab<<<g, 256, 0, st>>>(A, lda, sA, Bw, ldb, sB, C, ldc, sC, bias,
                                     M, Nn, Kk, alpha, accum);
}

static void G_ABT(hipStream_t st, const float* A, long lda, long sA,
                  const float* Bt, long ldb, long sB,
                  float* C, long ldc, long sC,
                  int M, int Nn, int Kk, float alpha, int batch)
{
    dim3 g(cdiv_(Nn, TN), cdiv_(M, TM), batch);
    sutra_gemm_abt_causal<<<g, 256, 0, st>>>(A, lda, sA, Bt, ldb, sB, C, ldc, sC,
                                             M, Nn, Kk, alpha);
}

extern "C" void kernel_launch(void* const* d_in, const int* in_sizes, int n_in,
                              void* d_out, int out_size, void* d_ws, size_t ws_size,
                              hipStream_t stream)
{
    const int*   x         = (const int*)  d_in[0];
    const float* emb       = (const float*)d_in[1];
    const float* pos_emb   = (const float*)d_in[2];
    const float* gru_w_ih  = (const float*)d_in[3];
    const float* gru_w_hh  = (const float*)d_in[4];
    const float* gru_b_ih  = (const float*)d_in[5];
    const float* gru_b_hh  = (const float*)d_in[6];
    const float* patch_ln_g= (const float*)d_in[7];
    const float* patch_ln_b= (const float*)d_in[8];
    const float* summ_w    = (const float*)d_in[9];
    const float* summ_b    = (const float*)d_in[10];
    const float* msg_w1    = (const float*)d_in[11];
    const float* msg_b1    = (const float*)d_in[12];
    const float* msg_w2    = (const float*)d_in[13];
    const float* msg_b2    = (const float*)d_in[14];
    const float* msg_ln_g  = (const float*)d_in[15];
    const float* msg_ln_b  = (const float*)d_in[16];
    const float* q_w = (const float*)d_in[17]; const float* q_b = (const float*)d_in[18];
    const float* k_w = (const float*)d_in[19]; const float* k_b = (const float*)d_in[20];
    const float* v_w = (const float*)d_in[21]; const float* v_b = (const float*)d_in[22];
    const float* o_w = (const float*)d_in[23]; const float* o_b = (const float*)d_in[24];
    const float* gc_w_ih   = (const float*)d_in[25];
    const float* gc_w_hh   = (const float*)d_in[26];
    const float* gc_b_ih   = (const float*)d_in[27];
    const float* gc_b_hh   = (const float*)d_in[28];
    const float* ar_ln_g   = (const float*)d_in[29];
    const float* ar_ln_b   = (const float*)d_in[30];
    const float* ar_qkv_w  = (const float*)d_in[31];
    const float* ar_qkv_b  = (const float*)d_in[32];
    const float* ar_out_w  = (const float*)d_in[33];
    const float* ar_out_b  = (const float*)d_in[34];
    const float* broad_w   = (const float*)d_in[35];
    const float* broad_b   = (const float*)d_in[36];
    const float* ln_g      = (const float*)d_in[37];
    const float* ln_b      = (const float*)d_in[38];
    const float* head_w    = (const float*)d_in[39];
    float* out = (float*)d_out;
    float* ws  = (float*)d_ws;

    // workspace layout (floats); scores+wi alias h0 (dead after GRU), qkv aliases gi
    const size_t OFF_LOCAL = 0;
    const size_t OFF_H0    = OFF_LOCAL + (size_t)MN_ * P_ * D_;
    const size_t OFF_SC    = OFF_H0;
    const size_t OFF_WI    = OFF_H0 + (size_t)B_ * N_ * N_;
    const size_t OFF_GI    = OFF_H0 + (size_t)MN_ * P_ * D_;
    const size_t OFF_QKV   = OFF_GI;
    const size_t OFF_GH    = OFF_GI + (size_t)MN_ * 1536;
    const size_t OFF_H     = OFF_GH + (size_t)MN_ * 1536;
    const size_t OFF_TMP   = OFF_H   + (size_t)MN_ * D_;
    const size_t OFF_U     = OFF_TMP + (size_t)MN_ * D_;   // also hn_
    const size_t OFF_VW    = OFF_U   + (size_t)MN_ * D_;   // also obuf
    const size_t OFF_G     = OFF_VW  + (size_t)MN_ * D_;   // also attn
    const size_t OFF_Q     = OFF_G   + (size_t)MN_ * D_;
    const size_t OFF_K     = OFF_Q   + (size_t)MN_ * D_;
    const size_t OFF_V     = OFF_K   + (size_t)MN_ * D_;
    const size_t OFF_Z     = OFF_V   + (size_t)MN_ * D_;
    const size_t OFF_COS   = OFF_Z   + (size_t)MN_ * D_;
    const size_t OFF_SIN   = OFF_COS + (size_t)N_ * 256;

    float* local  = ws + OFF_LOCAL;
    float* h0     = ws + OFF_H0;
    float* scores = ws + OFF_SC;
    float* wi     = ws + OFF_WI;
    float* gi     = ws + OFF_GI;
    float* qkv    = ws + OFF_QKV;
    float* gh     = ws + OFF_GH;
    float* h      = ws + OFF_H;
    float* tmp    = ws + OFF_TMP;
    float* U      = ws + OFF_U;
    float* hn_    = ws + OFF_U;
    float* Vw     = ws + OFF_VW;
    float* obuf   = ws + OFF_VW;
    float* Gb     = ws + OFF_G;
    float* attn   = ws + OFF_G;
    float* qb     = ws + OFF_Q;
    float* kb     = ws + OFF_K;
    float* vb     = ws + OFF_V;
    float* zbuf   = ws + OFF_Z;
    float* cosT   = ws + OFF_COS;
    float* sinT   = ws + OFF_SIN;
    float* broad  = tmp;

    (void)hipMemsetAsync(zbuf, 0, (size_t)MN_ * D_ * sizeof(float), stream);
    sutra_rope_table<<<cdiv_(N_ * 256, 256), 256, 0, stream>>>(cosT, sinT);
    sutra_embed<<<cdiv_(B_ * N_ * P_ * D_, 256), 256, 0, stream>>>(x, emb, pos_emb, h0);

    // ---- patch GRU scan (P=4 steps) ----
    for (int p = 0; p < P_; ++p) {
        G_AB(stream, h0 + (size_t)p * D_, 2048, 0, gru_w_ih, 1536, 0,
             gi, 1536, 0, gru_b_ih, MN_, 1536, 512, 1.f, 0, 1);
        const float* hp = (p == 0) ? zbuf : local + (size_t)(p - 1) * D_;
        long hs = (p == 0) ? 512 : 2048;
        G_AB(stream, hp, hs, 0, gru_w_hh, 1536, 0,
             gh, 1536, 0, gru_b_hh, MN_, 1536, 512, 1.f, 0, 1);
        sutra_gru_gate<<<cdiv_(MN_ * D_, 256), 256, 0, stream>>>(
            gi, gh, hp, hs, local + (size_t)p * D_, 2048, MN_);
    }
    sutra_layernorm<<<MN_ * P_, 256, 0, stream>>>(local, 512, local, 512,
                                                  patch_ln_g, patch_ln_b, MN_ * P_);
    sutra_patch_mean<<<cdiv_(MN_ * D_, 256), 256, 0, stream>>>(local, tmp);
    G_AB(stream, tmp, 512, 0, summ_w, 512, 0, h, 512, 0, summ_b, MN_, 512, 512, 1.f, 0, 1);

    const float sc_scale  = 0.04419417382415922f; // 1/sqrt(512)
    const float ar_scale  = 0.125f;               // 1/sqrt(64)

    for (int rnd = 0; rnd < ROUNDS_; ++rnd) {
        // message MLP: gelu(h@W1a + shift(h@W1b) + b1), mean over 5 windows, @W2
        G_AB(stream, h, 512, 0, msg_w1,              512, 0, U,  512, 0, nullptr, MN_, 512, 512, 1.f, 0, 1);
        G_AB(stream, h, 512, 0, msg_w1 + 512 * 512,  512, 0, Vw, 512, 0, nullptr, MN_, 512, 512, 1.f, 0, 1);
        sutra_msg_gelu<<<cdiv_(MN_ * D_, 256), 256, 0, stream>>>(U, Vw, msg_b1, Gb);
        G_AB(stream, Gb, 512, 0, msg_w2, 512, 0, wi, 1024, 0, msg_b2, MN_, 512, 512, 1.f, 0, 1);

        // sparse retrieval attention
        G_AB(stream, h, 512, 0, q_w, 512, 0, qb, 512, 0, q_b, MN_, 512, 512, 1.f, 0, 1);
        G_AB(stream, h, 512, 0, k_w, 512, 0, kb, 512, 0, k_b, MN_, 512, 512, 1.f, 0, 1);
        G_AB(stream, h, 512, 0, v_w, 512, 0, vb, 512, 0, v_b, MN_, 512, 512, 1.f, 0, 1);
        sutra_rotary<<<cdiv_(B_ * N_ * 256, 256), 256, 0, stream>>>(qb, cosT, sinT);
        sutra_rotary<<<cdiv_(B_ * N_ * 256, 256), 256, 0, stream>>>(kb, cosT, sinT);
        G_ABT(stream, qb, 512, (long)N_ * 512, kb, 512, (long)N_ * 512,
              scores, N_, (long)N_ * N_, N_, N_, 512, sc_scale, B_);
        sutra_topk_attend<<<MN_, 256, 0, stream>>>(scores, vb, attn);
        G_AB(stream, attn, 512, 0, o_w, 512, 0, wi + 512, 1024, 0, o_b, MN_, 512, 512, 1.f, 0, 1);

        // gated cell update + LN
        G_AB(stream, wi, 1024, 0, gc_w_ih, 1536, 0, gi, 1536, 0, gc_b_ih, MN_, 1536, 1024, 1.f, 0, 1);
        G_AB(stream, h,  512,  0, gc_w_hh, 1536, 0, gh, 1536, 0, gc_b_hh, MN_, 1536, 512,  1.f, 0, 1);
        sutra_gru_gate<<<cdiv_(MN_ * D_, 256), 256, 0, stream>>>(gi, gh, h, 512, tmp, 512, MN_);
        sutra_layernorm<<<MN_, 256, 0, stream>>>(tmp, 512, h, 512, msg_ln_g, msg_ln_b, MN_);

        // dense AR attention every second round
        if ((rnd & 1) == 1) {
            sutra_layernorm<<<MN_, 256, 0, stream>>>(h, 512, hn_, 512, ar_ln_g, ar_ln_b, MN_);
            G_AB(stream, hn_, 512, 0, ar_qkv_w, 1536, 0, qkv, 1536, 0, ar_qkv_b, MN_, 1536, 512, 1.f, 0, 1);
            for (int hh = 0; hh < H_; ++hh) {
                G_ABT(stream, qkv + hh * HD_,        1536, (long)N_ * 1536,
                              qkv + 512 + hh * HD_,  1536, (long)N_ * 1536,
                              scores, N_, (long)N_ * N_, N_, N_, HD_, ar_scale, B_);
                sutra_softmax_row<<<B_ * N_, 256, 0, stream>>>(scores, N_);
                G_AB(stream, scores, N_, (long)N_ * N_,
                             qkv + 1024 + hh * HD_, 1536, (long)N_ * 1536,
                             obuf + hh * HD_, 512, (long)N_ * 512, nullptr,
                     N_, HD_, N_, 1.f, 0, B_);
            }
            G_AB(stream, obuf, 512, 0, ar_out_w, 512, 0, h, 512, 0, ar_out_b,
                 MN_, 512, 512, 1.f, 1 /*accum: h = res + proj*/, 1);
        }
    }

    // broadcast + final LN + head
    G_AB(stream, h, 512, 0, broad_w, 512, 0, broad, 512, 0, broad_b, MN_, 512, 512, 1.f, 0, 1);
    sutra_final_add<<<cdiv_(B_ * T_ * D_, 256), 256, 0, stream>>>(local, broad);
    sutra_layernorm<<<B_ * T_, 256, 0, stream>>>(local, 512, local, 512, ln_g, ln_b, B_ * T_);
    G_AB(stream, local, 512, 0, head_w, 256, 0, out, 256, 0, nullptr, B_ * T_, 256, 512, 1.f, 0, 1);
}
